// MVSABlock_68676527063202
// MI455X (gfx1250) — matrix operand order repaired and checked by reading, not compiled, not run on hardware
//
#include <hip/hip_runtime.h>
#include <hip/hip_bf16.h>

#define NB     2
#define CDIM   256
#define DD     8
#define HWSZ   256
#define NTOK   2048          // D*H*W
#define NH     8
#define HD     32
#define MROWS  (NB * NTOK)   // 4096 tokens total

typedef __attribute__((ext_vector_type(16))) _Float16 v16h;
typedef __attribute__((ext_vector_type(8)))  float    v8f;

static __device__ inline v8f wmma_f16(v16h a, v16h b, v8f c) {
  return __builtin_amdgcn_wmma_f32_16x16x32_f16(false, a, false, b, (short)0, c, false, false);
}

// A (16xK tile of row-major [M][K], lda in halves): lane = M (mod 16),
// K(e) = e + (e>=8 ? 8 : 0) + (lane>=16 ? 8 : 0)   [ISA 16-bit A 16x32 layout]
static __device__ inline v16h load_A_frag(const _Float16* A, int lda, int m0, int k0, int lane) {
  const _Float16* p = A + (size_t)(m0 + (lane & 15)) * lda + k0 + ((lane & 16) ? 8 : 0);
  v16h a;
#pragma unroll
  for (int e = 0; e < 8; ++e) a[e] = p[e];
#pragma unroll
  for (int e = 0; e < 8; ++e) a[8 + e] = p[16 + e];
  return a;
}

// B (32x16 tile) loaded from B^T row-major [N][K]: lane = N (mod 16),
// K(e) = e + (e>=8 ? 8 : 0) + (lane>=16 ? 8 : 0)   [same K order as A]
static __device__ inline v16h load_B_frag(const _Float16* BT, int ldb, int n0, int k0, int lane) {
  const _Float16* p = BT + (size_t)(n0 + (lane & 15)) * ldb + k0 + ((lane & 16) ? 8 : 0);
  v16h b;
#pragma unroll
  for (int e = 0; e < 8; ++e) b[e] = p[e];
#pragma unroll
  for (int e = 0; e < 8; ++e) b[8 + e] = p[16 + e];
  return b;
}

// monotonic (ascending) 16-bit key for f16 ordering
static __device__ inline unsigned short f16key(_Float16 h) {
  unsigned short b = __builtin_bit_cast(unsigned short, h);
  return (b & 0x8000u) ? (unsigned short)(~b) : (unsigned short)(b | 0x8000u);
}

// -------- pooling (7-tap depth conv) + channel LayerNorm + x transpose --------
__global__ void pool_ln_kernel(const float* __restrict__ x, const float* __restrict__ y,
                               const float* __restrict__ lnw, const float* __restrict__ lnb,
                               _Float16* __restrict__ xl, _Float16* __restrict__ yl) {
  int bn = blockIdx.x;                 // b*NTOK + n
  int b = bn >> 11, n = bn & (NTOK - 1);
  int c = threadIdx.x;                 // channel
  int d = n >> 8, hw = n & (HWSZ - 1);
  const float w3 = 1.f / 3.f, w5 = 1.f / 5.f, w7 = 1.f / 7.f;
  const float wt[7] = {w7, w7 + w5, w7 + w5 + w3, w7 + w5 + w3, w7 + w5 + w3, w7 + w5, w7};
  const float* yb = y + ((size_t)b * CDIM + c) * (DD * HWSZ) + hw;
  float acc = 0.f;
#pragma unroll
  for (int dd = -3; dd <= 3; ++dd) {
    int dz = d + dd;
    if (dz >= 0 && dz < DD) acc += wt[dd + 3] * yb[dz * HWSZ];
  }
  __shared__ float s1[CDIM], s2[CDIM];
  s1[c] = acc; s2[c] = acc * acc;
  __syncthreads();
  for (int s = CDIM / 2; s > 0; s >>= 1) {
    if (c < s) { s1[c] += s1[c + s]; s2[c] += s2[c + s]; }
    __syncthreads();
  }
  float mu = s1[0] * (1.f / CDIM);
  float var = s2[0] * (1.f / CDIM) - mu * mu;
  float inv = rsqrtf(var + 1e-5f);
  yl[(size_t)bn * CDIM + c] = (_Float16)((acc - mu) * inv * lnw[c] + lnb[c]);
  xl[(size_t)bn * CDIM + c] = (_Float16)x[((size_t)b * CDIM + c) * NTOK + n];
}

// -------- weight transpose+convert to f16 [out][in] --------
__global__ void wconv_kernel(const float* __restrict__ Wq, const float* __restrict__ Wkv,
                             const float* __restrict__ Wproj,
                             _Float16* __restrict__ WqT, _Float16* __restrict__ WkvT,
                             _Float16* __restrict__ WprojT) {
  int gid = blockIdx.x * blockDim.x + threadIdx.x;   // 0..131071
  {
    int o = gid >> 8, i = gid & 255;                 // o < 512
    WkvT[(size_t)o * 256 + i] = (_Float16)Wkv[(size_t)i * 512 + o];
  }
  if (gid < 65536) {
    int o = gid >> 8, i = gid & 255;
    WqT[(size_t)o * 256 + i]    = (_Float16)Wq[(size_t)i * 256 + o];
    WprojT[(size_t)o * 256 + i] = (_Float16)Wproj[(size_t)i * 256 + o];
  }
}

// -------- q = (xl @ Wq + bq) * hd^-0.5, scattered to [b,h,n,d] --------
__global__ void q_gemm_kernel(const _Float16* __restrict__ xl, const _Float16* __restrict__ WqT,
                              const float* __restrict__ bq, _Float16* __restrict__ qh) {
  int lane = threadIdx.x & 31, wid = threadIdx.x >> 5;
  int tile = blockIdx.x * 8 + wid;
  if (tile >= (MROWS / 16) * (CDIM / 16)) return;    // wave-uniform
  int mt = tile >> 4, nt = tile & 15;
  v8f c = {};
  for (int k0 = 0; k0 < CDIM; k0 += 32) {
    v16h a  = load_A_frag(xl, CDIM, mt * 16, k0, lane);
    v16h bf = load_B_frag(WqT, CDIM, nt * 16, k0, lane);
    c = wmma_f16(a, bf, c);
  }
  const float scale = 0.17677669529663687f;          // 32^-0.5 folded into q
  int col = nt * 16 + (lane & 15);
  int h = col >> 5, dd = col & 31;
  float bias = bq[col];
  int ro = (lane & 16) ? 8 : 0;
#pragma unroll
  for (int i = 0; i < 8; ++i) {
    int row = mt * 16 + i + ro;
    int b = row >> 11, n = row & (NTOK - 1);
    qh[(((size_t)(b * NH + h)) * NTOK + n) * HD + dd] = (_Float16)((c[i] + bias) * scale);
  }
}

// -------- kv = yl @ Wkv + bkv; k -> [b,h,n,d], v -> transposed [b,h,d,n] --------
__global__ void kv_gemm_kernel(const _Float16* __restrict__ yl, const _Float16* __restrict__ WkvT,
                               const float* __restrict__ bkv,
                               _Float16* __restrict__ kh, _Float16* __restrict__ vt) {
  int lane = threadIdx.x & 31, wid = threadIdx.x >> 5;
  int tile = blockIdx.x * 8 + wid;
  if (tile >= (MROWS / 16) * (512 / 16)) return;
  int mt = tile >> 5, nt = tile & 31;
  v8f c = {};
  for (int k0 = 0; k0 < CDIM; k0 += 32) {
    v16h a  = load_A_frag(yl, CDIM, mt * 16, k0, lane);
    v16h bf = load_B_frag(WkvT, CDIM, nt * 16, k0, lane);
    c = wmma_f16(a, bf, c);
  }
  int col = nt * 16 + (lane & 15);
  float bias = bkv[col];
  int ro = (lane & 16) ? 8 : 0;
#pragma unroll
  for (int i = 0; i < 8; ++i) {
    int row = mt * 16 + i + ro;
    int b = row >> 11, n = row & (NTOK - 1);
    float v = c[i] + bias;
    if (col < 256) {
      int h = col >> 5, dd = col & 31;
      kh[(((size_t)(b * NH + h)) * NTOK + n) * HD + dd] = (_Float16)v;
    } else {
      int c2 = col - 256;
      int h = c2 >> 5, dd = c2 & 31;
      vt[(((size_t)(b * NH + h)) * HD + dd) * NTOK + n] = (_Float16)v;
    }
  }
}

// exact k-th-largest threshold via 2-pass radix select on f16 keys (per wave, uniform barriers)
static __device__ inline unsigned short radix_select(const _Float16* row, int n, int k,
                                                     int* hist, int lane) {
  for (int i = lane; i < 256; i += 32) hist[i] = 0;
  __syncthreads();
  for (int c = lane; c < n; c += 32) atomicAdd(&hist[f16key(row[c]) >> 8], 1);
  __syncthreads();
  int rem = k, hb = 0;
  for (int b = 255; b >= 0; --b) { int cnt = hist[b]; if (rem - cnt <= 0) { hb = b; break; } rem -= cnt; }
  __syncthreads();
  for (int i = lane; i < 256; i += 32) hist[i] = 0;
  __syncthreads();
  for (int c = lane; c < n; c += 32) {
    unsigned short key = f16key(row[c]);
    if ((key >> 8) == hb) atomicAdd(&hist[key & 0xFF], 1);
  }
  __syncthreads();
  int lb = 0;
  for (int b = 255; b >= 0; --b) { int cnt = hist[b]; if (rem - cnt <= 0) { lb = b; break; } rem -= cnt; }
  __syncthreads();
  return (unsigned short)((hb << 8) | lb);
}

// -------- fused attention: S tile in LDS, dual top-k softmax, combined P@V --------
__global__ void attn_kernel(const _Float16* __restrict__ q, const _Float16* __restrict__ kmat,
                            const _Float16* __restrict__ vt,
                            const float* __restrict__ a1p, const float* __restrict__ a2p,
                            _Float16* __restrict__ ao) {
  extern __shared__ char smem[];
  _Float16* sc = (_Float16*)smem;                                 // 16 x 2048 f16 = 64KB
  float* otile  = (float*)(smem + 16 * NTOK * 2);                 // 16 x 32 f32
  int*   hist   = (int*)(smem + 16 * NTOK * 2 + 512 * 4);         // 8 waves x 256 bins
  float* rowmax = (float*)(smem + 16 * NTOK * 2 + 512 * 4 + 8 * 256 * 4);
  unsigned short* thr1 = (unsigned short*)(rowmax + 16);
  unsigned short* thr2 = thr1 + 16;

  int lane = threadIdx.x & 31, wid = threadIdx.x >> 5;
  int blk = blockIdx.x;
  int mt = blk & 127, bh = blk >> 7;                              // bh = b*8 + h
  const _Float16* qbh = q    + (size_t)bh * NTOK * HD;
  const _Float16* kbh = kmat + (size_t)bh * NTOK * HD;
  const _Float16* vbh = vt   + (size_t)bh * HD * NTOK;
  int ro = (lane & 16) ? 8 : 0;

  // ---- S = Q K^T : one WMMA per 16x16 tile (K = hd = 32), 16 tiles per wave ----
  v16h afr = load_A_frag(qbh, HD, mt * 16, 0, lane);
#pragma unroll 1
  for (int t = 0; t < 16; ++t) {
    int n0 = (wid * 16 + t) * 16;
    v16h bfr = load_B_frag(kbh, HD, n0, 0, lane);
    v8f cz = {};
    v8f cc = wmma_f16(afr, bfr, cz);
    int col = n0 + (lane & 15);
#pragma unroll
    for (int i = 0; i < 8; ++i) sc[(size_t)(i + ro) * NTOK + col] = (_Float16)cc[i];
  }
  __syncthreads();

  // ---- per-row max + exact top-k thresholds (keep = N/2 and N/3) ----
  const int keep1 = NTOK / 2, keep2 = NTOK / 3;
  for (int rr = 0; rr < 2; ++rr) {                                // uniform across waves
    int r = wid * 2 + rr;
    const _Float16* row = sc + (size_t)r * NTOK;
    float mx = -3.0e38f;
    for (int c2 = lane; c2 < NTOK; c2 += 32) mx = fmaxf(mx, (float)row[c2]);
#pragma unroll
    for (int m2 = 16; m2 > 0; m2 >>= 1) mx = fmaxf(mx, __shfl_xor(mx, m2, 32));
    if (lane == 0) rowmax[r] = mx;
    unsigned short t1 = radix_select(row, NTOK, keep1, hist + wid * 256, lane);
    unsigned short t2 = radix_select(row, NTOK, keep2, hist + wid * 256, lane);
    if (lane == 0) { thr1[r] = t1; thr2[r] = t2; }
  }
  __syncthreads();

  // ---- combined masked softmax: P = a1*P1 + a2*P2, written back in place ----
  float a1v = a1p[0], a2v = a2p[0];
  for (int rr = 0; rr < 2; ++rr) {
    int r = wid * 2 + rr;
    _Float16* row = sc + (size_t)r * NTOK;
    float mx = rowmax[r];
    unsigned short t1 = thr1[r], t2 = thr2[r];
    float s1 = 0.f, s2 = 0.f;
    for (int c2 = lane; c2 < NTOK; c2 += 32) {
      _Float16 h = row[c2];
      unsigned short key = f16key(h);
      float e = __expf((float)h - mx);
      if (key >= t1) s1 += e;
      if (key >= t2) s2 += e;
    }
#pragma unroll
    for (int m2 = 16; m2 > 0; m2 >>= 1) { s1 += __shfl_xor(s1, m2, 32); s2 += __shfl_xor(s2, m2, 32); }
    float i1 = a1v / s1, i2 = a2v / s2;
    for (int c2 = lane; c2 < NTOK; c2 += 32) {
      _Float16 h = row[c2];
      unsigned short key = f16key(h);
      float e = __expf((float)h - mx);
      float p = (key >= t1 ? e * i1 : 0.f) + (key >= t2 ? e * i2 : 0.f);
      row[c2] = (_Float16)p;
    }
  }
  for (int e2 = threadIdx.x; e2 < 512; e2 += 256) otile[e2] = 0.f;
  __syncthreads();

  // ---- O = P @ V : K = 2048 split across 8 waves, 2 output col-tiles (hd=32) ----
  v8f acc0 = {}, acc1 = {};
  for (int kc = wid; kc < 64; kc += 8) {
    v16h pa = load_A_frag(sc, NTOK, 0, kc * 32, lane);
    v16h b0 = load_B_frag(vbh, NTOK, 0,  kc * 32, lane);
    v16h b1 = load_B_frag(vbh, NTOK, 16, kc * 32, lane);
    acc0 = wmma_f16(pa, b0, acc0);
    acc1 = wmma_f16(pa, b1, acc1);
  }
  int colx = lane & 15;
#pragma unroll
  for (int i = 0; i < 8; ++i) {
    atomicAdd(&otile[(i + ro) * 32 + colx], acc0[i]);
    atomicAdd(&otile[(i + ro) * 32 + 16 + colx], acc1[i]);
  }
  __syncthreads();

  int h = bh & 7, b = bh >> 3;
  for (int e2 = threadIdx.x; e2 < 512; e2 += 256) {
    int r = e2 >> 5, dd = e2 & 31;
    int n = mt * 16 + r;
    ao[((size_t)(b * NTOK + n)) * CDIM + h * HD + dd] = (_Float16)otile[e2];
  }
}

// -------- out = ao @ Wproj + bproj, stored transposed to (B,C,N) f32 --------
__global__ void proj_gemm_kernel(const _Float16* __restrict__ ao, const _Float16* __restrict__ WprojT,
                                 const float* __restrict__ bproj, float* __restrict__ out) {
  int lane = threadIdx.x & 31, wid = threadIdx.x >> 5;
  int tile = blockIdx.x * 8 + wid;
  if (tile >= (MROWS / 16) * (CDIM / 16)) return;
  int mt = tile >> 4, nt = tile & 15;
  v8f c = {};
  for (int k0 = 0; k0 < CDIM; k0 += 32) {
    v16h a  = load_A_frag(ao, CDIM, mt * 16, k0, lane);
    v16h bf = load_B_frag(WprojT, CDIM, nt * 16, k0, lane);
    c = wmma_f16(a, bf, c);
  }
  int col = nt * 16 + (lane & 15);
  float bias = bproj[col];
  int ro = (lane & 16) ? 8 : 0;
#pragma unroll
  for (int i = 0; i < 8; ++i) {
    int row = mt * 16 + i + ro;
    int b = row >> 11, n = row & (NTOK - 1);
    out[((size_t)b * CDIM + col) * NTOK + n] = c[i] + bias;
  }
}

extern "C" void kernel_launch(void* const* d_in, const int* in_sizes, int n_in,
                              void* d_out, int out_size, void* d_ws, size_t ws_size,
                              hipStream_t stream) {
  const float* x     = (const float*)d_in[0];
  const float* y     = (const float*)d_in[1];
  const float* lnw   = (const float*)d_in[2];
  const float* lnb   = (const float*)d_in[3];
  const float* Wq    = (const float*)d_in[4];
  const float* bq    = (const float*)d_in[5];
  const float* Wkv   = (const float*)d_in[6];
  const float* bkv   = (const float*)d_in[7];
  const float* Wproj = (const float*)d_in[8];
  const float* bproj = (const float*)d_in[9];
  const float* a1    = (const float*)d_in[10];
  const float* a2    = (const float*)d_in[11];
  float* out = (float*)d_out;

  char* ws = (char*)d_ws;
  auto take = [&](size_t bytes) { char* p = ws; ws += (bytes + 255) & ~(size_t)255; return p; };
  _Float16* xl     = (_Float16*)take((size_t)MROWS * CDIM * 2);
  _Float16* yl     = (_Float16*)take((size_t)MROWS * CDIM * 2);
  _Float16* WqT    = (_Float16*)take((size_t)256 * 256 * 2);
  _Float16* WkvT   = (_Float16*)take((size_t)512 * 256 * 2);
  _Float16* WprojT = (_Float16*)take((size_t)256 * 256 * 2);
  _Float16* qh     = (_Float16*)take((size_t)MROWS * CDIM * 2);
  _Float16* kh     = (_Float16*)take((size_t)MROWS * CDIM * 2);
  _Float16* vt     = (_Float16*)take((size_t)MROWS * CDIM * 2);
  _Float16* ao     = (_Float16*)take((size_t)MROWS * CDIM * 2);

  pool_ln_kernel<<<MROWS, 256, 0, stream>>>(x, y, lnw, lnb, xl, yl);
  wconv_kernel<<<512, 256, 0, stream>>>(Wq, Wkv, Wproj, WqT, WkvT, WprojT);
  q_gemm_kernel<<<(MROWS / 16) * (CDIM / 16) / 8, 256, 0, stream>>>(xl, WqT, bq, qh);
  kv_gemm_kernel<<<(MROWS / 16) * (512 / 16) / 8, 256, 0, stream>>>(yl, WkvT, bkv, kh, vt);

  size_t attn_lds = (size_t)16 * NTOK * 2 + 512 * 4 + 8 * 256 * 4 + 64 + 64;  // ~75.9 KB
  attn_kernel<<<NB * NH * (NTOK / 16), 256, attn_lds, stream>>>(qh, kh, vt, a1, a2, ao);

  proj_gemm_kernel<<<(MROWS / 16) * (CDIM / 16) / 8, 256, 0, stream>>>(ao, WprojT, bproj, out);
}
